// HeteroGNN_2138893713837
// MI455X (gfx1250) — compile-verified, matrix operand order checked by reading, not run on hardware
//
#include <hip/hip_runtime.h>
#include <hip/hip_bf16.h>

typedef __attribute__((ext_vector_type(16))) _Float16 v16h;
typedef __attribute__((ext_vector_type(8)))  float    v8f;

#define NN  20000
#define EE  320000
#define HH  4
#define CC  64
#define HC  256   // H*C
#define INF_ 128
#define OUTC 32

// ---------------------------------------------------------------------------
// Zero-fill (graph-capture-safe replacement for hipMemset)
// ---------------------------------------------------------------------------
__global__ void zero_kernel(float* __restrict__ p, long cnt) {
    long i = (long)blockIdx.x * blockDim.x + threadIdx.x;
    if (i < cnt) p[i] = 0.0f;
}

__global__ void relu_kernel(float* __restrict__ p, long cnt) {
    long i = (long)blockIdx.x * blockDim.x + threadIdx.x;
    if (i < cnt) p[i] = fmaxf(p[i], 0.0f);
}

// ---------------------------------------------------------------------------
// WMMA GEMM:  Hout[N,256] = X[N,K] @ W[K,256]   (one 64-col panel per blockIdx.y)
// block = 256 threads = 8 waves; each wave owns a 16-row x 64-col strip.
//
// B panel is staged into LDS *in WMMA fragment order*: for k-tile kt, column c,
// half-group hgrp (lanes 0-15 vs 16-31), the 16 halves a lane needs are stored
// contiguously at  Bs[(((kt*64 + c)*2 + hgrp) << 4) .. +15]  -> one aligned
// 32B LDS read per B fragment (2x ds_load_b128) instead of 16 ds_load_u16.
//
// ISA 16-bit A/B fragment K mapping per lane-half-group:
//   hgrp0 (lanes 0-15):  elems 0-7 -> K 0-7,  elems 8-15 -> K 16-23
//   hgrp1 (lanes 16-31): elems 0-7 -> K 8-15, elems 8-15 -> K 24-31
//   i.e.  e = (kin & 7) | ((kin & 16) ? 8 : 0),  hgrp = (kin >> 3) & 1
// ---------------------------------------------------------------------------
__global__ void gemm_wmma_kernel(const float* __restrict__ X,
                                 const float* __restrict__ W,
                                 float* __restrict__ Hout,
                                 int Nrows, int K) {
    __shared__ __attribute__((aligned(32))) _Float16 Bs[128 * 64];  // 16KB

    const int tid     = threadIdx.x;
    const int colBase = blockIdx.y * 64;

    // cooperative stage of W[:, colBase:colBase+64] as f16 in fragment order
    for (int idx = tid; idx < K * 64; idx += 256) {
        const int k = idx >> 6, c = idx & 63;
        const int kt = k >> 5, kin = k & 31;
        const int hg = (kin >> 3) & 1;
        const int e  = (kin & 7) | ((kin & 16) ? 8 : 0);
        Bs[((((kt * 64 + c) << 1) + hg) << 4) + e] =
            (_Float16)W[(size_t)k * HC + colBase + c];
    }
    __syncthreads();

    const int wave = tid >> 5;
    const int lane = tid & 31;
    const int rowBase = blockIdx.x * 128 + wave * 16;
    const int hgrp = (lane < 16) ? 0 : 1;
    const int kgrp = hgrp << 3;
    int row = rowBase + (lane & 15);
    if (row >= Nrows) row = Nrows - 1;          // clamp: keep EXEC uniform
    const float* xrow = X + (size_t)row * K;
    const int ncol = lane & 15;

    v8f acc[4] = {};

    const int nkt = K >> 5;
    for (int kt = 0; kt < nkt; ++kt) {
        const int kk = kt << 5;
        // A fragment: 4x float4 global loads + f32->f16 converts
        const float4 p0 = *(const float4*)(xrow + kk + kgrp);
        const float4 p1 = *(const float4*)(xrow + kk + kgrp + 4);
        const float4 p2 = *(const float4*)(xrow + kk + kgrp + 16);
        const float4 p3 = *(const float4*)(xrow + kk + kgrp + 20);
        v16h a;
        a[0]  = (_Float16)p0.x; a[1]  = (_Float16)p0.y;
        a[2]  = (_Float16)p0.z; a[3]  = (_Float16)p0.w;
        a[4]  = (_Float16)p1.x; a[5]  = (_Float16)p1.y;
        a[6]  = (_Float16)p1.z; a[7]  = (_Float16)p1.w;
        a[8]  = (_Float16)p2.x; a[9]  = (_Float16)p2.y;
        a[10] = (_Float16)p2.z; a[11] = (_Float16)p2.w;
        a[12] = (_Float16)p3.x; a[13] = (_Float16)p3.y;
        a[14] = (_Float16)p3.z; a[15] = (_Float16)p3.w;

        #pragma unroll
        for (int t = 0; t < 4; ++t) {
            const v16h b = *(const v16h*)
                &Bs[(((((kt * 64) + t * 16 + ncol) << 1) + hgrp) << 4)];
            acc[t] = __builtin_amdgcn_wmma_f32_16x16x32_f16(
                         false, a, false, b, (short)0, acc[t], false, false);
        }
    }

    // D store: row = rowBase + r + 8*hgrp, col = colBase + t*16 + (lane&15)
    const int rofs = hgrp << 3;
    #pragma unroll
    for (int t = 0; t < 4; ++t) {
        const int c = colBase + t * 16 + ncol;
        #pragma unroll
        for (int r = 0; r < 8; ++r) {
            const int rr = rowBase + rofs + r;
            if (rr < Nrows) Hout[(size_t)rr * HC + c] = acc[t][r];
        }
    }
}

// ---------------------------------------------------------------------------
// Attention vectors: asv[n,h] = sum_c h[n,h,c]*a_s[h,c] ; same for adv.
// One wave per node; lane -> 8 channels; 8-lane xor-reduce per head.
// ---------------------------------------------------------------------------
__global__ void avec_kernel(const float* __restrict__ Hm,
                            const float* __restrict__ a_s,
                            const float* __restrict__ a_d,
                            float* __restrict__ asv, float* __restrict__ adv,
                            int Nn) {
    const int wave = threadIdx.x >> 5, lane = threadIdx.x & 31;
    const int n = blockIdx.x * 8 + wave;
    if (n >= Nn) return;
    const float4* hv4 = (const float4*)(Hm + (size_t)n * HC + lane * 8);
    const float4* as4 = (const float4*)(a_s + lane * 8);
    const float4* ad4 = (const float4*)(a_d + lane * 8);
    const float4 h0 = hv4[0], h1 = hv4[1];
    const float4 s0 = as4[0], s1 = as4[1];
    const float4 d0 = ad4[0], d1 = ad4[1];
    float ss = h0.x*s0.x + h0.y*s0.y + h0.z*s0.z + h0.w*s0.w
             + h1.x*s1.x + h1.y*s1.y + h1.z*s1.z + h1.w*s1.w;
    float sd = h0.x*d0.x + h0.y*d0.y + h0.z*d0.z + h0.w*d0.w
             + h1.x*d1.x + h1.y*d1.y + h1.z*d1.z + h1.w*d1.w;
    #pragma unroll
    for (int m = 1; m < 8; m <<= 1) {
        ss += __shfl_xor(ss, m, 32);
        sd += __shfl_xor(sd, m, 32);
    }
    if ((lane & 7) == 0) {
        const int h = lane >> 3;
        asv[n * HH + h] = ss;
        adv[n * HH + h] = sd;
    }
}

// ---------------------------------------------------------------------------
// Edge scatter: grid-stride waves over edges (incl. self loops). Unshifted
// segment softmax num/denominator with f32 global atomics. Next iteration's
// h[src] row is prefetched (global_prefetch_b8) to hide gather latency.
// lane -> head = lane>>3, channels lane*8 .. lane*8+7.
// ---------------------------------------------------------------------------
__global__ void edge_kernel(const int* __restrict__ edge, int E_, int Nn,
                            const float* __restrict__ Hm,
                            const float* __restrict__ asv,
                            const float* __restrict__ adv,
                            float* __restrict__ z, float* __restrict__ num,
                            int nWaves) {
    const int wave = threadIdx.x >> 5, lane = threadIdx.x & 31;
    const int gw = blockIdx.x * 8 + wave;
    const long total = (long)E_ + Nn;
    const int head = lane >> 3;

    for (long ei = gw; ei < total; ei += nWaves) {
        int s, d;
        if (ei < E_) { s = edge[ei]; d = edge[E_ + ei]; }
        else         { s = d = (int)(ei - E_); }

        const long en = ei + nWaves;                 // prefetch next src row
        if (en < total) {
            const int sn = (en < E_) ? edge[en] : (int)(en - E_);
            __builtin_prefetch(Hm + (size_t)sn * HC + lane * 8, 0, 1);
        }

        float e = asv[s * HH + head] + adv[d * HH + head];
        e = (e > 0.0f) ? e : 0.2f * e;               // leaky_relu(0.2)
        const float w = __expf(e);

        if ((lane & 7) == 0) atomicAdd(&z[d * HH + head], w);

        const float4* hs = (const float4*)(Hm + (size_t)s * HC + lane * 8);
        const float4 h0 = hs[0], h1 = hs[1];
        float* nd = num + (size_t)d * HC + lane * 8;
        atomicAdd(&nd[0], w * h0.x); atomicAdd(&nd[1], w * h0.y);
        atomicAdd(&nd[2], w * h0.z); atomicAdd(&nd[3], w * h0.w);
        atomicAdd(&nd[4], w * h1.x); atomicAdd(&nd[5], w * h1.y);
        atomicAdd(&nd[6], w * h1.z); atomicAdd(&nd[7], w * h1.w);
    }
}

// ---------------------------------------------------------------------------
// Per-node normalize, head mean, +bias, /4 type-mean, accumulate into xacc.
// ---------------------------------------------------------------------------
__global__ void finalize_kernel(const float* __restrict__ num,
                                const float* __restrict__ z,
                                const float* __restrict__ b,
                                float* __restrict__ xacc, int Nn) {
    const long idx = (long)blockIdx.x * blockDim.x + threadIdx.x;
    if (idx >= (long)Nn * CC) return;
    const int n = (int)(idx >> 6), c = (int)(idx & 63);
    float acc = 0.0f;
    #pragma unroll
    for (int h = 0; h < HH; ++h)
        acc += num[(size_t)n * HC + h * CC + c] / z[n * HH + h];
    xacc[idx] += 0.25f * (acc * 0.25f + b[c]);   // head mean (/4), type mean (/4)
}

// ---------------------------------------------------------------------------
// Output linear: out[n, o] = x[n,:] @ lin_W[:,o] + lin_b[o]  (one wave / node)
// ---------------------------------------------------------------------------
__global__ void linout_kernel(const float* __restrict__ Xf,
                              const float* __restrict__ Wl,
                              const float* __restrict__ bl,
                              float* __restrict__ out, int Nn) {
    const int wave = threadIdx.x >> 5, lane = threadIdx.x & 31;
    const int n = blockIdx.x * 8 + wave;
    if (n >= Nn) return;
    float acc = bl[lane];
    const float* xr = Xf + (size_t)n * CC;
    #pragma unroll 8
    for (int c = 0; c < CC; ++c) acc += xr[c] * Wl[c * OUTC + lane];
    out[(size_t)n * OUTC + lane] = acc;
}

// ---------------------------------------------------------------------------
extern "C" void kernel_launch(void* const* d_in, const int* in_sizes, int n_in,
                              void* d_out, int out_size, void* d_ws, size_t ws_size,
                              hipStream_t stream) {
    const float* x_in  = (const float*)d_in[0];
    const int*   edges[4] = { (const int*)d_in[1], (const int*)d_in[2],
                              (const int*)d_in[3], (const int*)d_in[4] };
    const float* W0  = (const float*)d_in[5];
    const float* as0 = (const float*)d_in[6];
    const float* ad0 = (const float*)d_in[7];
    const float* b0  = (const float*)d_in[8];
    const float* W1  = (const float*)d_in[9];
    const float* as1 = (const float*)d_in[10];
    const float* ad1 = (const float*)d_in[11];
    const float* b1  = (const float*)d_in[12];
    const float* linW = (const float*)d_in[13];
    const float* linb = (const float*)d_in[14];

    float* ws  = (float*)d_ws;
    float* h   = ws;                       // N*256
    float* num = h   + (size_t)NN * HC;    // N*256
    float* z   = num + (size_t)NN * HC;    // N*4
    float* asv = z   + (size_t)NN * HH;    // N*4
    float* adv = asv + (size_t)NN * HH;    // N*4
    float* xB  = adv + (size_t)NN * HH;    // N*64  (layer0 out / layer1 in)
    float* xA  = xB  + (size_t)NN * CC;    // N*64  (layer1 out)

    const int gemmBlocksX = (NN + 127) / 128;              // 157
    const int edgeBlocks  = 8192;                          // grid-stride waves
    const int edgeWaves   = edgeBlocks * 8;
    const int nodeWaveBlocks = (NN + 7) / 8;               // 2500
    const long ncElems  = (long)NN * CC;
    const long nhcElems = (long)NN * HC;

    auto zero = [&](float* p, long cnt) {
        zero_kernel<<<(int)((cnt + 255) / 256), 256, 0, stream>>>(p, cnt);
    };

    for (int layer = 0; layer < 2; ++layer) {
        const float* xin  = (layer == 0) ? x_in : xB;
        float*       xacc = (layer == 0) ? xB   : xA;
        const int    K    = (layer == 0) ? INF_ : CC;
        const float* Wl   = (layer == 0) ? W0   : W1;
        const float* asl  = (layer == 0) ? as0  : as1;
        const float* adl  = (layer == 0) ? ad0  : ad1;
        const float* bl   = (layer == 0) ? b0   : b1;

        zero(xacc, ncElems);
        for (int t = 0; t < 4; ++t) {
            gemm_wmma_kernel<<<dim3(gemmBlocksX, 4), 256, 0, stream>>>(
                xin, Wl + (size_t)t * K * HC, h, NN, K);
            avec_kernel<<<nodeWaveBlocks, 256, 0, stream>>>(
                h, asl + t * HC, adl + t * HC, asv, adv, NN);
            zero(num, nhcElems);
            zero(z, (long)NN * HH);
            edge_kernel<<<edgeBlocks, 256, 0, stream>>>(
                edges[t], EE, NN, h, asv, adv, z, num, edgeWaves);
            finalize_kernel<<<(int)((ncElems + 255) / 256), 256, 0, stream>>>(
                num, z, bl + t * CC, xacc, NN);
        }
        relu_kernel<<<(int)((ncElems + 255) / 256), 256, 0, stream>>>(xacc, ncElems);
    }

    linout_kernel<<<nodeWaveBlocks, 256, 0, stream>>>(
        xA, linW, linb, (float*)d_out, NN);
}
